// OscillatorLayer_5068061409911
// MI455X (gfx1250) — matrix operand order repaired and checked by reading, not compiled, not run on hardware
//
#include <hip/hip_runtime.h>
#include <hip/hip_bf16.h>

// ---------------------------------------------------------------------------
// OscillatorLayer on MI455X (gfx1250, wave32)
//   phase 1: force = x @ W^T + b  -> bf16 WMMA (v_wmma_f32_16x16x32_bf16),
//            LDS-tiled 128x128 blocks, 8 waves * (4x2) 16x16 tiles each.
//   phase 2: per-column linear 2x2 recurrence over B, parallelized with a
//            chunked affine scan (M is constant per column), fused osc+clip.
// ---------------------------------------------------------------------------

typedef __attribute__((ext_vector_type(16))) __bf16 v16bf;
typedef __attribute__((ext_vector_type(8)))  __bf16 v8bf;
typedef __attribute__((ext_vector_type(8)))  float  v8f;
typedef __attribute__((ext_vector_type(4)))  float  v4f;

#define TWO_PI_F 6.2831853071795864769f
#define DT_F     0.01f

#define BM 128
#define BN 128
#define BK 32
#define LDK 40     // padded K-stride (bf16 elems): 80B row stride -> conflict-free-ish
#define NCHUNK 64  // scan chunks

// ---------------------------------------------------------------------------
// GEMM: out[B,N] = x[B,D] * W[N,D]^T + bias[N]
// ---------------------------------------------------------------------------
__global__ __launch_bounds__(256)
void osc_gemm_wmma(const float* __restrict__ x, const float* __restrict__ W,
                   const float* __restrict__ bias, float* __restrict__ out,
                   int D, int N)
{
    __shared__ __align__(16) __bf16 As[BM * LDK];   // x tile  [row M][k]
    __shared__ __align__(16) __bf16 Bs[BN * LDK];   // W tile  [row n][k]

    const int t    = threadIdx.x;
    const int lane = t & 31;
    const int lr   = lane & 15;   // 0..15
    const int hl   = lane >> 4;   // 0 or 1 (lane half)
    const int wave = t >> 5;      // 0..7
    const int wm   = wave >> 2;   // 0..1 : 64-row slice
    const int wn   = wave & 3;    // 0..3 : 32-col slice

    const int m0 = blockIdx.y * BM;
    const int n0 = blockIdx.x * BN;

    // staging: thread -> (row 0..127, 16-wide half of K)
    const int srow = t >> 1;
    const int scol = (t & 1) * 16;

    const float* xp = x + (size_t)(m0 + srow) * D + scol;
    const float* wp = W + (size_t)(n0 + srow) * D + scol;

    v8f acc[4][2];
#pragma unroll
    for (int i = 0; i < 4; ++i)
#pragma unroll
        for (int j = 0; j < 2; ++j)
            acc[i][j] = (v8f){0.f,0.f,0.f,0.f,0.f,0.f,0.f,0.f};

    for (int k0 = 0; k0 < D; k0 += BK) {
        // prefetch next K tile into cache (global_prefetch_b8)
        __builtin_prefetch(xp + BK, 0, 3);
        __builtin_prefetch(wp + BK, 0, 3);

        // global fp32 -> bf16 -> LDS
        float xv[16], wv[16];
#pragma unroll
        for (int q = 0; q < 4; ++q) {
            *(v4f*)(xv + 4*q) = *(const v4f*)(xp + 4*q);
            *(v4f*)(wv + 4*q) = *(const v4f*)(wp + 4*q);
        }
        v8bf a0, a1, b0, b1;
#pragma unroll
        for (int q = 0; q < 8; ++q) {
            a0[q] = (__bf16)xv[q];     a1[q] = (__bf16)xv[q + 8];
            b0[q] = (__bf16)wv[q];     b1[q] = (__bf16)wv[q + 8];
        }
        *(v8bf*)(&As[srow * LDK + scol    ]) = a0;
        *(v8bf*)(&As[srow * LDK + scol + 8]) = a1;
        *(v8bf*)(&Bs[srow * LDK + scol    ]) = b0;
        *(v8bf*)(&Bs[srow * LDK + scol + 8]) = b1;
        __syncthreads();

        // A fragment (16x32 bf16): lane half 0 holds K = 0..7 & 16..23,
        //                          lane half 1 holds K = 8..15 & 24..31
        v16bf af[4];
#pragma unroll
        for (int i = 0; i < 4; ++i) {
            const __bf16* ap = &As[(wm*64 + i*16 + lr) * LDK + hl*8];
            v8bf lo = *(const v8bf*)ap;
            v8bf hi = *(const v8bf*)(ap + 16);
            af[i] = __builtin_shufflevector(lo, hi,
                      0,1,2,3,4,5,6,7,8,9,10,11,12,13,14,15);
        }
        // B fragment (32x16 bf16): lanes 0-15 hold K=0..7 & 16..23, lanes 16-31 K=8..15 & 24..31
        v16bf bfr[2];
#pragma unroll
        for (int j = 0; j < 2; ++j) {
            const __bf16* bp = &Bs[(wn*32 + j*16 + lr) * LDK + hl*8];
            v8bf lo = *(const v8bf*)bp;
            v8bf hi = *(const v8bf*)(bp + 16);
            bfr[j] = __builtin_shufflevector(lo, hi,
                      0,1,2,3,4,5,6,7,8,9,10,11,12,13,14,15);
        }

#pragma unroll
        for (int i = 0; i < 4; ++i)
#pragma unroll
            for (int j = 0; j < 2; ++j)
                acc[i][j] = __builtin_amdgcn_wmma_f32_16x16x32_bf16(
                    /*neg_a=*/false, af[i], /*neg_b=*/false, bfr[j],
                    /*c_mod=*/(short)0, acc[i][j],
                    /*reuse_a=*/false, /*reuse_b=*/false);
        __syncthreads();

        xp += BK; wp += BK;
    }

    // epilogue: C layout -> VGPR vi, lane: row = 8*hl + vi, col = lr
#pragma unroll
    for (int j = 0; j < 2; ++j) {
        const int col = n0 + wn*32 + j*16 + lr;
        const float bv = bias[col];
#pragma unroll
        for (int i = 0; i < 4; ++i) {
            const int rbase = m0 + wm*64 + i*16 + hl*8;
#pragma unroll
            for (int vi = 0; vi < 8; ++vi)
                out[(size_t)(rbase + vi) * N + col] = acc[i][j][vi] + bv;
        }
    }
}

// ---------------------------------------------------------------------------
// Scan pass 1: per (chunk c, column n), run recurrence from zero state,
// record chunk-end (state, vel).
// ---------------------------------------------------------------------------
__global__ __launch_bounds__(256)
void scan_partial(const float* __restrict__ force, float* __restrict__ ends,
                  const float* __restrict__ freq, const float* __restrict__ damp,
                  int N, int L, int C)
{
    const int n = blockIdx.x * blockDim.x + threadIdx.x;
    const int c = blockIdx.y;
    if (n >= N) return;
    const float om = TWO_PI_F * freq[n];
    const float os = om * om;
    const float dp = damp[n];
    float s = 0.f, v = 0.f;
    const float* fp = force + (size_t)c * L * N + n;
    for (int b = 0; b < L; ++b) {
        const float f = fp[(size_t)b * N];
        const float a = -os * s - dp * v + f;
        v += a * DT_F;
        s += v * DT_F;
    }
    ends[(size_t)c * N + n]               = s;
    ends[(size_t)C * N + (size_t)c * N + n] = v;
}

// ---------------------------------------------------------------------------
// Scan pass 2: per column, z_{c+1} = M^L * z_c + e_c  (M constant per column)
// ---------------------------------------------------------------------------
__global__ __launch_bounds__(256)
void scan_combine(const float* __restrict__ ends, float* __restrict__ inits,
                  const float* __restrict__ freq, const float* __restrict__ damp,
                  int N, int L, int C)
{
    const int n = blockIdx.x * blockDim.x + threadIdx.x;
    if (n >= N) return;
    const float om = TWO_PI_F * freq[n];
    const float os = om * om;
    const float dp = damp[n];
    // one-step map M
    const float m00 = 1.f - os * DT_F * DT_F;
    const float m01 = DT_F * (1.f - dp * DT_F);
    const float m10 = -os * DT_F;
    const float m11 = 1.f - dp * DT_F;
    // P = M^L
    float p00 = 1.f, p01 = 0.f, p10 = 0.f, p11 = 1.f;
    for (int i = 0; i < L; ++i) {
        const float q00 = m00 * p00 + m01 * p10;
        const float q01 = m00 * p01 + m01 * p11;
        const float q10 = m10 * p00 + m11 * p10;
        const float q11 = m10 * p01 + m11 * p11;
        p00 = q00; p01 = q01; p10 = q10; p11 = q11;
    }
    float s = 0.f, v = 0.f;
    for (int c = 0; c < C; ++c) {
        inits[(size_t)c * N + n]                 = s;
        inits[(size_t)C * N + (size_t)c * N + n] = v;
        const float es = ends[(size_t)c * N + n];
        const float ev = ends[(size_t)C * N + (size_t)c * N + n];
        const float ns = p00 * s + p01 * v + es;
        const float nv = p10 * s + p11 * v + ev;
        s = ns; v = nv;
    }
}

// ---------------------------------------------------------------------------
// Scan pass 3: replay each chunk from its exact init, fuse osc + clip,
// in place over the force buffer (== d_out).
// ---------------------------------------------------------------------------
__global__ __launch_bounds__(256)
void scan_final(float* __restrict__ out, const float* __restrict__ inits,
                const float* __restrict__ freq, const float* __restrict__ damp,
                const float* __restrict__ amp, const float* __restrict__ phase,
                int N, int L, int C)
{
    const int n = blockIdx.x * blockDim.x + threadIdx.x;
    const int c = blockIdx.y;
    if (n >= N) return;
    const float om = TWO_PI_F * freq[n];
    const float os = om * om;
    const float dp = damp[n];
    const float osc = amp[n] * __sinf(om * DT_F + phase[n]);
    float s = inits[(size_t)c * N + n];
    float v = inits[(size_t)C * N + (size_t)c * N + n];
    float* p = out + (size_t)c * L * N + n;
    for (int b = 0; b < L; ++b) {
        const float f = p[(size_t)b * N];
        const float a = -os * s - dp * v + f;
        v += a * DT_F;
        s += v * DT_F;
        float r = s + osc;
        r = fminf(1.f, fmaxf(-1.f, r));
        p[(size_t)b * N] = r;
    }
}

// ---------------------------------------------------------------------------
extern "C" void kernel_launch(void* const* d_in, const int* in_sizes, int n_in,
                              void* d_out, int out_size, void* d_ws, size_t ws_size,
                              hipStream_t stream)
{
    const float* x     = (const float*)d_in[0];
    const float* W     = (const float*)d_in[1];
    const float* bias  = (const float*)d_in[2];
    const float* amp   = (const float*)d_in[3];
    const float* freq  = (const float*)d_in[4];
    const float* phase = (const float*)d_in[5];
    const float* damp  = (const float*)d_in[6];
    float* out = (float*)d_out;
    float* ws  = (float*)d_ws;

    const int N = in_sizes[2];            // 4096
    const int D = in_sizes[1] / N;        // 2048
    const int B = in_sizes[0] / D;        // 4096

    // Phase 1: GEMM into d_out (serves as the force buffer).
    dim3 gemm_grid(N / BN, B / BM);
    osc_gemm_wmma<<<gemm_grid, 256, 0, stream>>>(x, W, bias, out, D, N);

    // Phase 2: chunked affine scan over the batch dimension.
    const int C = NCHUNK;
    const int L = B / C;
    float* ends  = ws;                        // 2*C*N floats
    float* inits = ws + (size_t)2 * C * N;    // 2*C*N floats

    dim3 scan_grid((N + 255) / 256, C);
    scan_partial<<<scan_grid, 256, 0, stream>>>(out, ends, freq, damp, N, L, C);
    scan_combine<<<dim3((N + 255) / 256), 256, 0, stream>>>(ends, inits, freq, damp, N, L, C);
    scan_final  <<<scan_grid, 256, 0, stream>>>(out, inits, freq, damp, amp, phase, N, L, C);
}
